// ConvGNN_30159260352847
// MI455X (gfx1250) — matrix operand order repaired and checked by reading, not compiled, not run on hardware
//
#include <hip/hip_runtime.h>

typedef __attribute__((ext_vector_type(16))) _Float16 v16h;
typedef __attribute__((ext_vector_type(8)))  _Float16 v8h;
typedef __attribute__((ext_vector_type(8)))  float    v8f;

#define N_NODES 100000
#define N_EDGES 800000
#define NBLK_N  ((N_NODES + 255) / 256)   // 391

// ---------------------------------------------------------------------------
// Pre-swizzle W2 (64x64) and W3 (64x256) into f16 WMMA B-fragment layout.
// B fragment (32x16 f16, K x N): lane l holds column n = l&15, half = l>>4,
// element j -> K = 32*c + 8*half + (j&7) + 16*(j>>3)  (c = K-chunk). Fragment f = t*2 + c.
// Stored fragment-major: frag[f][lane][j] contiguous -> 32B/lane loads.
// ---------------------------------------------------------------------------
__global__ __launch_bounds__(256) void prep_frags_kernel(
    const float* __restrict__ W2, const float* __restrict__ W3,
    _Float16* __restrict__ w2f, _Float16* __restrict__ w3f)
{
    int tid = blockIdx.x * 256 + threadIdx.x;
    if (tid < 8 * 512) {
        int f = tid >> 9, r = tid & 511;
        int lane = r >> 4, j = r & 15;
        int t = f >> 1, c = f & 1;
        int k = 32 * c + 8 * (lane >> 4) + (j & 7) + 16 * (j >> 3);
        int n = 16 * t + (lane & 15);
        w2f[tid] = (_Float16)W2[k * 64 + n];
    } else {
        int tid2 = tid - 8 * 512;
        int f = tid2 >> 9, r = tid2 & 511;
        int lane = r >> 4, j = r & 15;
        int t = f >> 1, c = f & 1;
        int k = 32 * c + 8 * (lane >> 4) + (j & 7) + 16 * (j >> 3);
        int n = 16 * t + (lane & 15);
        w3f[tid2] = (_Float16)W3[k * 256 + n];
    }
}

__global__ __launch_bounds__(256) void init_h0_kernel(
    const float* __restrict__ x, float* __restrict__ h0, int n)
{
    int v = blockIdx.x * 256 + threadIdx.x;
    if (v >= n) return;
    for (int i = 0; i < 8; ++i)  h0[(size_t)v * 16 + i] = x[(size_t)v * 8 + i];
    for (int i = 8; i < 16; ++i) h0[(size_t)v * 16 + i] = 0.0f;
}

__global__ __launch_bounds__(256) void zero_f_kernel(float* __restrict__ p, int n)
{
    int i = blockIdx.x * 256 + threadIdx.x;
    if (i < n) p[i] = 0.0f;
}

__global__ __launch_bounds__(256) void zero_i_kernel(int* __restrict__ p, int n)
{
    int i = blockIdx.x * 256 + threadIdx.x;
    if (i < n) p[i] = 0;
}

// ---------------------------------------------------------------------------
// CSR-by-dst build: histogram -> two-level exclusive scan -> rank scatter.
// dst[] is static across all 3 rounds, so this runs once per call.
// ---------------------------------------------------------------------------
__global__ __launch_bounds__(256) void hist_kernel(
    const int* __restrict__ ei, int* __restrict__ cnt)
{
    int e = blockIdx.x * 256 + threadIdx.x;
    atomicAdd(&cnt[ei[e * 2 + 1]], 1);
}

__global__ __launch_bounds__(256) void scan1_kernel(
    const int* __restrict__ in, int* __restrict__ out,
    int* __restrict__ bsum, int n)
{
    __shared__ int s[256];
    int i = blockIdx.x * 256 + threadIdx.x;
    int v = (i < n) ? in[i] : 0;
    s[threadIdx.x] = v;
    __syncthreads();
    for (int d = 1; d < 256; d <<= 1) {
        int t = (threadIdx.x >= (unsigned)d) ? s[threadIdx.x - d] : 0;
        __syncthreads();
        s[threadIdx.x] += t;
        __syncthreads();
    }
    if (i < n) out[i] = s[threadIdx.x] - v;            // exclusive within block
    if (threadIdx.x == 255) bsum[blockIdx.x] = s[255]; // block total
}

__global__ __launch_bounds__(512) void scan2_kernel(int* __restrict__ b, int n)
{
    __shared__ int s[512];
    int v = (threadIdx.x < (unsigned)n) ? b[threadIdx.x] : 0;
    s[threadIdx.x] = v;
    __syncthreads();
    for (int d = 1; d < 512; d <<= 1) {
        int t = (threadIdx.x >= (unsigned)d) ? s[threadIdx.x - d] : 0;
        __syncthreads();
        s[threadIdx.x] += t;
        __syncthreads();
    }
    if (threadIdx.x < (unsigned)n) b[threadIdx.x] = s[threadIdx.x] - v; // exclusive
}

__global__ __launch_bounds__(256) void scan3_kernel(
    int* __restrict__ offs, const int* __restrict__ bsum,
    int* __restrict__ cursor, int n)
{
    int i = blockIdx.x * 256 + threadIdx.x;
    if (i < n) {
        int o = offs[i] + bsum[blockIdx.x];
        offs[i]   = o;
        cursor[i] = o;
    }
}

__global__ __launch_bounds__(256) void scatter_kernel(
    const int* __restrict__ ei, int* __restrict__ cursor,
    int* __restrict__ eperm, int* __restrict__ srcs)
{
    int e = blockIdx.x * 256 + threadIdx.x;
    int d = ei[e * 2 + 1];
    int p = atomicAdd(&cursor[d], 1);
    eperm[p] = e;
    srcs[p]  = ei[e * 2 + 0];
}

// ---------------------------------------------------------------------------
// Edge MLP: per wave, 16 edges. L1 in VALU -> LDS; L2/L3 via WMMA f16.
// grid = E/128 blocks of 256 threads (8 waves), exact cover.
// ---------------------------------------------------------------------------
__global__ __launch_bounds__(256) void edge_mlp_kernel(
    const float* __restrict__ edge_attr,
    const float* __restrict__ W1, const float* __restrict__ b1,
    const float* __restrict__ b2, const float* __restrict__ b3,
    const _Float16* __restrict__ w2f, const _Float16* __restrict__ w3f,
    _Float16* __restrict__ wedge)
{
    __shared__ _Float16 lds_all[8][16 * 72];   // 18,432 B / block
    const int lane = threadIdx.x & 31;
    const int wave = threadIdx.x >> 5;
    _Float16* X = lds_all[wave];
    const int m    = lane & 15;
    const int hf   = lane >> 4;
    const int nsel = lane & 15;
    const int ebase = blockIdx.x * 128 + wave * 16;

    // ---- Layer 1: X1[m][k] = relu(b1[k] + ea·W1[:,k]), lane does 32 k's ----
    {
        float4 ea = *(const float4*)(edge_attr + (size_t)(ebase + m) * 4);
        const int k0 = 32 * hf;
        for (int kk = 0; kk < 32; ++kk) {
            int k = k0 + kk;
            float s = b1[k] + ea.x * W1[0 * 64 + k] + ea.y * W1[1 * 64 + k]
                            + ea.z * W1[2 * 64 + k] + ea.w * W1[3 * 64 + k];
            X[m * 72 + k] = (_Float16)fmaxf(s, 0.0f);
        }
    }
    __syncthreads();

    // ---- A fragments of X1 (16x32 f16 layout: two 16B chunks per lane) ----
    v16h a0, a1;
    {
        union { v16h v; v8h h[2]; } u;
        const _Float16* row = X + m * 72;
        u.h[0] = *(const v8h*)(row + 0  + 8 * hf);
        u.h[1] = *(const v8h*)(row + 16 + 8 * hf);
        a0 = u.v;
        u.h[0] = *(const v8h*)(row + 32 + 8 * hf);
        u.h[1] = *(const v8h*)(row + 48 + 8 * hf);
        a1 = u.v;
    }
    __syncthreads();

    // ---- Layer 2: (16x64)@(64x64), 4 N-tiles x 2 WMMAs; relu -> X2 LDS ----
    for (int t = 0; t < 4; ++t) {
        float bv = b2[t * 16 + nsel];
        v8f c;
        for (int r = 0; r < 8; ++r) c[r] = bv;
        v16h bf0 = *(const v16h*)(w2f + (t * 2 + 0) * 512 + lane * 16);
        v16h bf1 = *(const v16h*)(w2f + (t * 2 + 1) * 512 + lane * 16);
        c = __builtin_amdgcn_wmma_f32_16x16x32_f16(false, a0, false, bf0,
                                                   (short)0, c, false, false);
        c = __builtin_amdgcn_wmma_f32_16x16x32_f16(false, a1, false, bf1,
                                                   (short)0, c, false, false);
        // D layout: vgpr r, lane l -> M = r + 8*(l>>4), N = l&15
        for (int r = 0; r < 8; ++r) {
            int mr = r + 8 * hf;
            X[mr * 72 + t * 16 + nsel] = (_Float16)fmaxf(c[r], 0.0f);
        }
    }
    __syncthreads();

    // ---- A fragments of X2 ----
    v16h a2, a3;
    {
        union { v16h v; v8h h[2]; } u;
        const _Float16* row = X + m * 72;
        u.h[0] = *(const v8h*)(row + 0  + 8 * hf);
        u.h[1] = *(const v8h*)(row + 16 + 8 * hf);
        a2 = u.v;
        u.h[0] = *(const v8h*)(row + 32 + 8 * hf);
        u.h[1] = *(const v8h*)(row + 48 + 8 * hf);
        a3 = u.v;
    }

    // ---- Layer 3: (16x64)@(64x256), 16 N-tiles x 2 WMMAs -> Wedge f16 ----
    for (int t = 0; t < 16; ++t) {
        float bv = b3[t * 16 + nsel];
        v8f c;
        for (int r = 0; r < 8; ++r) c[r] = bv;
        v16h bf0 = *(const v16h*)(w3f + (t * 2 + 0) * 512 + lane * 16);
        v16h bf1 = *(const v16h*)(w3f + (t * 2 + 1) * 512 + lane * 16);
        c = __builtin_amdgcn_wmma_f32_16x16x32_f16(false, a2, false, bf0,
                                                   (short)0, c, false, false);
        c = __builtin_amdgcn_wmma_f32_16x16x32_f16(false, a3, false, bf1,
                                                   (short)0, c, false, false);
        // column block t of D == Wedge[e][t][0..15]; NT: 410MB stream > L2
        for (int r = 0; r < 8; ++r) {
            int e = ebase + r + 8 * hf;
            __builtin_nontemporal_store((_Float16)c[r],
                wedge + (size_t)e * 256 + t * 16 + nsel);
        }
    }
}

// ---------------------------------------------------------------------------
// Fused message + aggregate + node update, atomic-free via dst-CSR gather.
// Half-wave (16 lanes) per node; lane = output channel o.
//   h_out[n][o] = bias[o] + sum_i h[n][i]*root[i][o]
//              + sum_{p in CSR(n)} sum_i h[srcs[p]][i] * Wedge[eperm[p]][i][o]
// grid = N*16/256 = 6250 blocks, exact cover.
// ---------------------------------------------------------------------------
__global__ __launch_bounds__(256) void msg_update_kernel(
    const float* __restrict__ h_in, const _Float16* __restrict__ wedge,
    const int* __restrict__ offs, const int* __restrict__ cnt,
    const int* __restrict__ eperm, const int* __restrict__ srcs,
    const float* __restrict__ root, const float* __restrict__ bias,
    float* __restrict__ h_out)
{
    int gt   = blockIdx.x * 256 + threadIdx.x;
    int node = gt >> 4;
    int o    = gt & 15;

    float hv[16];
    for (int i = 0; i < 16; ++i) hv[i] = h_in[(size_t)node * 16 + i];
    float s = bias[o];
    for (int i = 0; i < 16; ++i) s += hv[i] * root[i * 16 + o];

    int start = offs[node];
    int deg   = cnt[node];
    for (int p = 0; p < deg; ++p) {
        int eid = eperm[start + p];
        int src = srcs[start + p];
        float sv[16];
        for (int i = 0; i < 16; ++i) sv[i] = h_in[(size_t)src * 16 + i];
        const _Float16* w = wedge + (size_t)eid * 256 + o;
        float acc = 0.0f;
        for (int i = 0; i < 16; ++i)
            acc += sv[i] * (float)__builtin_nontemporal_load(w + i * 16);
        s += acc;
    }
    h_out[(size_t)node * 16 + o] = s;
}

__global__ __launch_bounds__(256) void readout_kernel(
    const float* __restrict__ h, const float* __restrict__ h0,
    const float* __restrict__ Wi1, const float* __restrict__ bi1,
    const float* __restrict__ Wi2, const float* __restrict__ bi2,
    const float* __restrict__ Wj1, const float* __restrict__ bj1,
    const float* __restrict__ Wj2, const float* __restrict__ bj2,
    float* __restrict__ out, int n)
{
    __shared__ float red[256];
    int v = blockIdx.x * 256 + threadIdx.x;
    float contrib = 0.0f;
    if (v < n) {
        float hv[16], h0v[16];
        for (int i = 0; i < 16; ++i) {
            hv[i]  = h[(size_t)v * 16 + i];
            h0v[i] = h0[(size_t)v * 16 + i];
        }
        float g = bi2[0], val = bj2[0];
        for (int r = 0; r < 128; ++r) {
            float s = bi1[r];
            for (int i = 0; i < 16; ++i) s += hv[i]  * Wi1[i * 128 + r];
            for (int i = 0; i < 16; ++i) s += h0v[i] * Wi1[(16 + i) * 128 + r];
            g += fmaxf(s, 0.0f) * Wi2[r];
            float sv = bj1[r];
            for (int i = 0; i < 16; ++i) sv += hv[i] * Wj1[i * 128 + r];
            val += fmaxf(sv, 0.0f) * Wj2[r];
        }
        float gate = 1.0f / (1.0f + __expf(-g));
        contrib = gate * val;
    }
    red[threadIdx.x] = contrib;
    __syncthreads();
    for (int s = 128; s > 0; s >>= 1) {
        if (threadIdx.x < (unsigned)s) red[threadIdx.x] += red[threadIdx.x + s];
        __syncthreads();
    }
    if (threadIdx.x == 0) atomicAdd(out, red[0]);
}

// ---------------------------------------------------------------------------
extern "C" void kernel_launch(void* const* d_in, const int* in_sizes, int n_in,
                              void* d_out, int out_size, void* d_ws, size_t ws_size,
                              hipStream_t stream)
{
    (void)in_sizes; (void)n_in; (void)out_size; (void)ws_size;
    const float* x   = (const float*)d_in[0];
    const int*   ei  = (const int*)d_in[1];
    const float* ea  = (const float*)d_in[2];
    const float* W1  = (const float*)d_in[3];
    const float* b1  = (const float*)d_in[4];
    const float* W2  = (const float*)d_in[5];
    const float* b2  = (const float*)d_in[6];
    const float* W3  = (const float*)d_in[7];
    const float* b3  = (const float*)d_in[8];
    const float* root= (const float*)d_in[9];
    const float* bias= (const float*)d_in[10];
    const float* Wi1 = (const float*)d_in[11];
    const float* bi1 = (const float*)d_in[12];
    const float* Wi2 = (const float*)d_in[13];
    const float* bi2 = (const float*)d_in[14];
    const float* Wj1 = (const float*)d_in[15];
    const float* bj1 = (const float*)d_in[16];
    const float* Wj2 = (const float*)d_in[17];
    const float* bj2 = (const float*)d_in[18];
    float* out = (float*)d_out;

    char* ws = (char*)d_ws;
    size_t off = 0;
    _Float16* wedge = (_Float16*)(ws + off); off += (size_t)N_EDGES * 256 * 2; // 409.6 MB
    float* hA   = (float*)(ws + off); off += (size_t)N_NODES * 16 * 4;          // h0
    float* hB   = (float*)(ws + off); off += (size_t)N_NODES * 16 * 4;
    float* hC   = (float*)(ws + off); off += (size_t)N_NODES * 16 * 4;
    int* cnt    = (int*)(ws + off); off += (size_t)(N_NODES + 64) * 4;
    int* offs   = (int*)(ws + off); off += (size_t)(N_NODES + 64) * 4;
    int* cursor = (int*)(ws + off); off += (size_t)(N_NODES + 64) * 4;
    int* bsum   = (int*)(ws + off); off += 512 * 4;
    int* eperm  = (int*)(ws + off); off += (size_t)N_EDGES * 4;
    int* srcs   = (int*)(ws + off); off += (size_t)N_EDGES * 4;
    _Float16* w2f = (_Float16*)(ws + off); off += 8  * 512 * 2;
    _Float16* w3f = (_Float16*)(ws + off); off += 32 * 512 * 2;

    // one-time per call: weight swizzle, h0, output zero
    prep_frags_kernel<<<80, 256, 0, stream>>>(W2, W3, w2f, w3f);
    init_h0_kernel<<<NBLK_N, 256, 0, stream>>>(x, hA, N_NODES);
    zero_f_kernel<<<1, 256, 0, stream>>>(out, 1);

    // dst-CSR build (dst is identical across all 3 rounds)
    zero_i_kernel<<<NBLK_N, 256, 0, stream>>>(cnt, N_NODES);
    hist_kernel<<<N_EDGES / 256, 256, 0, stream>>>(ei, cnt);
    scan1_kernel<<<NBLK_N, 256, 0, stream>>>(cnt, offs, bsum, N_NODES);
    scan2_kernel<<<1, 512, 0, stream>>>(bsum, NBLK_N);
    scan3_kernel<<<NBLK_N, 256, 0, stream>>>(offs, bsum, cursor, N_NODES);
    scatter_kernel<<<N_EDGES / 256, 256, 0, stream>>>(ei, cursor, eperm, srcs);

    // WMMA edge MLP -> Wedge (f16)
    edge_mlp_kernel<<<N_EDGES / 128, 256, 0, stream>>>(ea, W1, b1, b2, b3,
                                                       w2f, w3f, wedge);

    // 3 atomic-free fused message+update rounds (ping-pong h)
    const float* hin = hA;
    float* houts[3] = { hB, hC, hB };
    for (int it = 0; it < 3; ++it) {
        msg_update_kernel<<<(N_NODES * 16) / 256, 256, 0, stream>>>(
            hin, wedge, offs, cnt, eperm, srcs, root, bias, houts[it]);
        hin = houts[it];
    }

    readout_kernel<<<NBLK_N, 256, 0, stream>>>(
        hin, hA, Wi1, bi1, Wi2, bi2, Wj1, bj1, Wj2, bj2, out, N_NODES);
}
